// PCN_52450140619426
// MI455X (gfx1250) — matrix operand order repaired and checked by reading, not compiled, not run on hardware
//
#include <hip/hip_runtime.h>
#include <hip/hip_bf16.h>

// PCN inference loop for MI455X (gfx1250, wave32).
// Big GEMMs -> v_wmma_f32_16x16x32_f16 (f16 in, f32 accum), 32x64 wave tiles
// (8 wmma per k-step, 1.5 b128 loads per wmma). All data L2-resident; weights
// converted/transposed once per launch.

#define B_SZ   8192
#define N0     784      // input dim
#define N1     1024
#define N2     512
#define N3     10
#define N0P    832      // 784 padded to multiple of 64 (N and K padding for W1 GEMMs)
#define LRC    0.001f
#define CYCLES 60

typedef __attribute__((ext_vector_type(16))) _Float16 v16h;
typedef __attribute__((ext_vector_type(8)))  float    v8f;

union FragH { v16h v; uint4 q[2]; };
union AccF  { v8f  v; float f[8]; };

__device__ __forceinline__ float sigmoidf(float x) { return 1.f / (1.f + __expf(-x)); }

// A-matrix 16x32 f16 fragment (ISA 7.12.2): lane m<16 holds row m, K {0..7,16..23};
// lane m+16 holds row m, K {8..15,24..31}. Two 16B loads per lane.
__device__ __forceinline__ v16h load_a_frag(const _Float16* __restrict__ base, int ld,
                                            int kt, int lane) {
  FragH f;
  const _Float16* p = base + (size_t)(lane & 15) * ld + kt + ((lane & 16) ? 8 : 0);
  f.q[0] = *(const uint4*)(p);
  f.q[1] = *(const uint4*)(p + 16);
  return f.v;
}

// B-matrix 32x16 f16 fragment: lane n<16 holds column n, K {0..7,16..23}; lane n+16 holds
// column n, K {8..15,24..31}. With B stored (N,K) row-major (NT GEMM) this is two 16B
// loads per lane.
__device__ __forceinline__ v16h load_b_frag(const _Float16* __restrict__ base, int ld,
                                            int kt, int lane) {
  FragH f;
  const _Float16* p = base + (size_t)(lane & 15) * ld + kt + ((lane & 16) ? 8 : 0);
  f.q[0] = *(const uint4*)(p);
  f.q[1] = *(const uint4*)(p + 16);
  return f.v;
}

// C(M x Np) = A(M x Kp) * B(Np x Kp)^T with fused epilogues.
// Block: 256 threads = 8 waves stacked in M; wave tile 32(M) x 64(N).
// kind 0: e0 = input - C     -> EH (f16, ld N0P, pad cols zeroed)
// kind 1: e1 = s1 - C        -> EH (f16)
// kind 2: s1 += LR*(-e1 + sigp(s1)*C)   (e1 read from EH f16)
// kind 3: s2 += LR*(-e2 + sigp(s2)*C)   (e2 read from EF f32)
__global__ __launch_bounds__(256)
void gemm_nt_kernel(const _Float16* __restrict__ A, const _Float16* __restrict__ Bm,
                    int Kp, int Nact,
                    float* S, int ldS,
                    _Float16* EH, int ldEH,
                    const float* EF, int ldEF,
                    int kind) {
  const int wave   = threadIdx.x >> 5;
  const int lane   = threadIdx.x & 31;
  const int m_wave = blockIdx.y * 256 + wave * 32;   // 8 waves stacked in M
  const int n0     = blockIdx.x * 64;

  v8f acc[2][4] = {};
  const _Float16* Ar[2];
  const _Float16* Br[4];
#pragma unroll
  for (int ms = 0; ms < 2; ms++) Ar[ms] = A + (size_t)(m_wave + 16 * ms) * Kp;
#pragma unroll
  for (int ns = 0; ns < 4; ns++) Br[ns] = Bm + (size_t)(n0 + 16 * ns) * Kp;

  for (int kt = 0; kt < Kp; kt += 32) {
    v16h a0 = load_a_frag(Ar[0], Kp, kt, lane);
    v16h a1 = load_a_frag(Ar[1], Kp, kt, lane);
    v16h b0 = load_b_frag(Br[0], Kp, kt, lane);
    v16h b1 = load_b_frag(Br[1], Kp, kt, lane);
    v16h b2 = load_b_frag(Br[2], Kp, kt, lane);
    v16h b3 = load_b_frag(Br[3], Kp, kt, lane);
    acc[0][0] = __builtin_amdgcn_wmma_f32_16x16x32_f16(false, a0, false, b0, (short)0, acc[0][0], false, false);
    acc[0][1] = __builtin_amdgcn_wmma_f32_16x16x32_f16(false, a0, false, b1, (short)0, acc[0][1], false, false);
    acc[0][2] = __builtin_amdgcn_wmma_f32_16x16x32_f16(false, a0, false, b2, (short)0, acc[0][2], false, false);
    acc[0][3] = __builtin_amdgcn_wmma_f32_16x16x32_f16(false, a0, false, b3, (short)0, acc[0][3], false, false);
    acc[1][0] = __builtin_amdgcn_wmma_f32_16x16x32_f16(false, a1, false, b0, (short)0, acc[1][0], false, false);
    acc[1][1] = __builtin_amdgcn_wmma_f32_16x16x32_f16(false, a1, false, b1, (short)0, acc[1][1], false, false);
    acc[1][2] = __builtin_amdgcn_wmma_f32_16x16x32_f16(false, a1, false, b2, (short)0, acc[1][2], false, false);
    acc[1][3] = __builtin_amdgcn_wmma_f32_16x16x32_f16(false, a1, false, b3, (short)0, acc[1][3], false, false);
  }

  // C/D layout: VGPR i, lane l -> M = i + (l>=16 ? 8 : 0), N = l&15
  const int lm = (lane & 16) ? 8 : 0;
  const int ln = lane & 15;
#pragma unroll
  for (int ms = 0; ms < 2; ms++)
#pragma unroll
    for (int ns = 0; ns < 4; ns++) {
      AccF u; u.v = acc[ms][ns];
#pragma unroll
      for (int i = 0; i < 8; i++) {
        int m = m_wave + ms * 16 + lm + i;
        int n = n0 + ns * 16 + ln;
        float a = u.f[i];
        if (kind <= 1) {
          float e = 0.f;
          if (n < Nact) e = S[(size_t)m * ldS + n] - a;
          EH[(size_t)m * ldEH + n] = (_Float16)e;      // pad cols -> 0 every cycle
        } else if (n < Nact) {
          float s  = S[(size_t)m * ldS + n];
          float sg = sigmoidf(s);
          float e  = (kind == 2) ? (float)EH[(size_t)m * ldEH + n]
                                 : EF[(size_t)m * ldEF + n];
          S[(size_t)m * ldS + n] = s + LRC * (-e + sg * (1.f - sg) * a);
        }
      }
    }
}

// Convert weights to f16 once per launch: W1h (832x1024, rows>=784 zero),
// W1Th (1024x832, k>=784 zero), W2h (1024x512), W2Th (512x1024).
__global__ void prep_weights_kernel(const float* __restrict__ W1, const float* __restrict__ W2,
                                    _Float16* W1h, _Float16* W1Th,
                                    _Float16* W2h, _Float16* W2Th) {
  int idx = blockIdx.x * 256 + threadIdx.x;
  const int c0 = N0P * N1;
  const int c1 = c0 + N1 * N0P;
  const int c2 = c1 + N1 * N2;
  const int c3 = c2 + N2 * N1;
  if (idx < c0) {
    int n = idx / N1, k = idx % N1;
    W1h[idx] = (n < N0) ? (_Float16)W1[n * N1 + k] : (_Float16)0.f;
  } else if (idx < c1) {
    int t = idx - c0; int n = t / N0P, k = t % N0P;
    W1Th[t] = (k < N0) ? (_Float16)W1[k * N1 + n] : (_Float16)0.f;
  } else if (idx < c2) {
    int t = idx - c1;
    W2h[t] = (_Float16)W2[t];
  } else if (idx < c3) {
    int t = idx - c2; int n = t / N1, k = t % N1;
    W2Th[t] = (_Float16)W2[k * N2 + n];
  }
}

__global__ void copy_states_kernel(const float* __restrict__ s1, const float* __restrict__ s2,
                                   const float* __restrict__ s3,
                                   float* s1w, float* s2w, float* s3w) {
  int idx = blockIdx.x * 256 + threadIdx.x;
  const int c1 = B_SZ * N1, c2 = c1 + B_SZ * N2, c3 = c2 + B_SZ * N3;
  if (idx < c1)      s1w[idx]      = s1[idx];
  else if (idx < c2) s2w[idx - c1] = s2[idx - c1];
  else if (idx < c3) s3w[idx - c2] = s3[idx - c2];
}

__global__ void act_kernel(const float* __restrict__ s1w, const float* __restrict__ s2w,
                           const float* __restrict__ s3w,
                           _Float16* a1h, _Float16* a2h, float* a3f) {
  int idx = blockIdx.x * 256 + threadIdx.x;
  const int c1 = B_SZ * N1, c2 = c1 + B_SZ * N2, c3 = c2 + B_SZ * N3;
  if (idx < c1)      a1h[idx]      = (_Float16)sigmoidf(s1w[idx]);
  else if (idx < c2) a2h[idx - c1] = (_Float16)sigmoidf(s2w[idx - c1]);
  else if (idx < c3) a3f[idx - c2] = sigmoidf(s3w[idx - c2]);
}

// e2 = s2 - sigmoid(s3) @ W3^T   (K=10, scalar f32)
__global__ void err2_kernel(const float* __restrict__ a3f, const float* __restrict__ W3,
                            const float* __restrict__ s2w, float* __restrict__ e2f) {
  int idx = blockIdx.x * 256 + threadIdx.x;   // B_SZ*N2 threads
  int m = idx >> 9, n = idx & 511;
  float acc = 0.f;
#pragma unroll
  for (int k = 0; k < N3; k++) acc += a3f[m * N3 + k] * W3[n * N3 + k];
  e2f[idx] = s2w[idx] - acc;
}

// s3 += LR * sigp(s3) * (e2 @ W3)   (N=10, scalar f32)
__global__ void upd3_kernel(const float* __restrict__ e2f, const float* __restrict__ W3,
                            float* __restrict__ s3w) {
  int idx = blockIdx.x * 256 + threadIdx.x;
  if (idx >= B_SZ * N3) return;
  int m = idx / N3, n = idx % N3;
  float g = 0.f;
  for (int k = 0; k < N2; k++) g += e2f[m * N2 + k] * W3[k * N3 + n];
  float s  = s3w[idx];
  float sg = sigmoidf(s);
  s3w[idx] = s + LRC * sg * (1.f - sg) * g;
}

// d_out = concat(s3 flat, p0_last = input - e0)
__global__ void finalize_kernel(const float* __restrict__ s3w, const float* __restrict__ input,
                                const _Float16* __restrict__ e0h, float* __restrict__ out) {
  int idx = blockIdx.x * 256 + threadIdx.x;
  const int c1 = B_SZ * N3;
  const int c2 = c1 + B_SZ * N0;
  if (idx < c1) out[idx] = s3w[idx];
  else if (idx < c2) {
    int t = idx - c1;
    int m = t / N0, n = t % N0;
    out[idx] = input[t] - (float)e0h[(size_t)m * N0P + n];
  }
}

extern "C" void kernel_launch(void* const* d_in, const int* in_sizes, int n_in,
                              void* d_out, int out_size, void* d_ws, size_t ws_size,
                              hipStream_t stream) {
  const float* input = (const float*)d_in[0];
  const float* s1 = (const float*)d_in[1];
  const float* s2 = (const float*)d_in[2];
  const float* s3 = (const float*)d_in[3];
  const float* W1 = (const float*)d_in[4];
  const float* W2 = (const float*)d_in[5];
  const float* W3 = (const float*)d_in[6];

  // Workspace layout (~130 MB total)
  char* ws = (char*)d_ws;
  size_t off = 0;
  auto alloc = [&](size_t bytes) -> char* {
    char* p = ws + off; off += (bytes + 255) & ~(size_t)255; return p;
  };
  _Float16* W1h  = (_Float16*)alloc((size_t)N0P * N1 * 2);
  _Float16* W1Th = (_Float16*)alloc((size_t)N1 * N0P * 2);
  _Float16* W2h  = (_Float16*)alloc((size_t)N1 * N2 * 2);
  _Float16* W2Th = (_Float16*)alloc((size_t)N2 * N1 * 2);
  _Float16* a1h  = (_Float16*)alloc((size_t)B_SZ * N1 * 2);
  _Float16* a2h  = (_Float16*)alloc((size_t)B_SZ * N2 * 2);
  float*    a3f  = (float*)   alloc((size_t)B_SZ * N3 * 4);
  _Float16* e0h  = (_Float16*)alloc((size_t)B_SZ * N0P * 2);
  _Float16* e1h  = (_Float16*)alloc((size_t)B_SZ * N1 * 2);
  float*    e2f  = (float*)   alloc((size_t)B_SZ * N2 * 4);
  float*    s1w  = (float*)   alloc((size_t)B_SZ * N1 * 4);
  float*    s2w  = (float*)   alloc((size_t)B_SZ * N2 * 4);
  float*    s3w  = (float*)   alloc((size_t)B_SZ * N3 * 4);

  const int prep_total = N0P * N1 + N1 * N0P + N1 * N2 + N2 * N1;
  prep_weights_kernel<<<(prep_total + 255) / 256, 256, 0, stream>>>(W1, W2, W1h, W1Th, W2h, W2Th);
  const int st_total = B_SZ * (N1 + N2 + N3);
  copy_states_kernel<<<(st_total + 255) / 256, 256, 0, stream>>>(s1, s2, s3, s1w, s2w, s3w);

  dim3 blk(256);
  for (int c = 0; c < CYCLES; c++) {
    act_kernel<<<(st_total + 255) / 256, 256, 0, stream>>>(s1w, s2w, s3w, a1h, a2h, a3f);
    // err0: e0 = input - sigmoid(s1) @ W1^T   (M=8192, N=784->832, K=1024)
    gemm_nt_kernel<<<dim3(N0P / 64, B_SZ / 256), blk, 0, stream>>>(
        a1h, W1h, N1, N0, (float*)input, N0, e0h, N0P, nullptr, 0, 0);
    // err1: e1 = s1 - sigmoid(s2) @ W2^T      (N=1024, K=512)
    gemm_nt_kernel<<<dim3(N1 / 64, B_SZ / 256), blk, 0, stream>>>(
        a2h, W2h, N2, N1, s1w, N1, e1h, N1, nullptr, 0, 1);
    err2_kernel<<<(B_SZ * N2) / 256, 256, 0, stream>>>(a3f, W3, s2w, e2f);
    // upd1: s1 += LR*(-e1 + sigp(s1)*(e0 @ W1))   (N=1024, K=832)
    gemm_nt_kernel<<<dim3(N1 / 64, B_SZ / 256), blk, 0, stream>>>(
        e0h, W1Th, N0P, N1, s1w, N1, e1h, N1, nullptr, 0, 2);
    // upd2: s2 += LR*(-e2 + sigp(s2)*(e1 @ W2))   (N=512, K=1024)
    gemm_nt_kernel<<<dim3(N2 / 64, B_SZ / 256), blk, 0, stream>>>(
        e1h, W2Th, N1, N2, s2w, N2, nullptr, 0, e2f, N2, 3);
    upd3_kernel<<<(B_SZ * N3 + 255) / 256, 256, 0, stream>>>(e2f, W3, s3w);
  }

  const int fin_total = B_SZ * N3 + B_SZ * N0;
  finalize_kernel<<<(fin_total + 255) / 256, 256, 0, stream>>>(s3w, input, e0h, (float*)d_out);
}
